// Llama4DecoderLayer_20684562497549
// MI455X (gfx1250) — compile-verified, matrix operand order checked by reading, not run on hardware
//
#include <hip/hip_runtime.h>
#include <stdint.h>

#define T_TOK 16384
#define DMODEL 1024
#define IMODEL 2048
#define NEXP   16

#define BM 128
#define BN 64
#define BK 32
#define LDST 40   // padded LDS row stride in u16 (80B, 16B-aligned, bank-spread)

typedef __attribute__((ext_vector_type(16))) __bf16 v16bf;
typedef __attribute__((ext_vector_type(8)))  float  v8f;
typedef int vi4 __attribute__((vector_size(16)));   // matches builtin prototype

#ifndef __has_builtin
#define __has_builtin(x) 0
#endif
#if __has_builtin(__builtin_amdgcn_global_load_async_to_lds_b128) && \
    __has_builtin(__builtin_amdgcn_s_wait_asynccnt)
#define HAVE_ASYNC_LDS 1
#else
#define HAVE_ASYNC_LDS 0
#endif

__device__ __forceinline__ uint16_t f32_bf16(float f) {
  uint32_t x = __float_as_uint(f);
  uint32_t r = x + 0x7FFFu + ((x >> 16) & 1u);
  return (uint16_t)(r >> 16);
}
__device__ __forceinline__ uint32_t pack_bf16x2(float lo, float hi) {
  return (uint32_t)f32_bf16(lo) | ((uint32_t)f32_bf16(hi) << 16);
}
__device__ __forceinline__ v8f zero8() {
  v8f z;
  #pragma unroll
  for (int i = 0; i < 8; ++i) z[i] = 0.f;
  return z;
}

// 16-byte global -> LDS copy. Async (ASYNCcnt-tracked, no VGPR round trip)
// when the gfx1250 builtins exist; otherwise synchronous via VGPRs.
__device__ __forceinline__ void copy16_g2lds(const void* gsrc, void* ldst) {
#if HAVE_ASYNC_LDS
  __builtin_amdgcn_global_load_async_to_lds_b128(
      (__attribute__((address_space(1))) vi4*)(uintptr_t)gsrc,
      (__attribute__((address_space(3))) vi4*)(uint32_t)(uintptr_t)ldst,
      0, 0);
#else
  *reinterpret_cast<uint4*>(ldst) = *reinterpret_cast<const uint4*>(gsrc);
#endif
}
#if HAVE_ASYNC_LDS
#define WAIT_ASYNC(n) __builtin_amdgcn_s_wait_asynccnt(n)
#else
#define WAIT_ASYNC(n)
#endif

union FragBF { v16bf v; uint32_t u[8]; uint4 q[2]; };

// --------------------------------------------------------------------------
// f32 -> bf16 bulk conversion (weights), 8 elems/thread.
// --------------------------------------------------------------------------
__global__ void __launch_bounds__(256)
cvt_bf16_kernel(const float* __restrict__ src, uint16_t* __restrict__ dst, int n) {
  int i = (blockIdx.x * 256 + threadIdx.x) * 8;
  if (i + 8 <= n) {
    float4 a = *reinterpret_cast<const float4*>(src + i);
    float4 b = *reinterpret_cast<const float4*>(src + i + 4);
    uint4 o;
    o.x = pack_bf16x2(a.x, a.y);
    o.y = pack_bf16x2(a.z, a.w);
    o.z = pack_bf16x2(b.x, b.y);
    o.w = pack_bf16x2(b.z, b.w);
    *reinterpret_cast<uint4*>(dst + i) = o;
  }
}

// --------------------------------------------------------------------------
// Router: one wave32 per token. 16 logits, top-1 + sigmoid, scatter into
// per-expert token list, emit xb (bf16 x) and xs (bf16 w*x).
// --------------------------------------------------------------------------
__global__ void __launch_bounds__(256)
router_kernel(const float* __restrict__ x, const float* __restrict__ wr,
              uint16_t* __restrict__ xb, uint16_t* __restrict__ xs,
              int* __restrict__ counts, int* __restrict__ lists) {
  int wave = threadIdx.x >> 5;
  int lane = threadIdx.x & 31;
  int t = blockIdx.x * 8 + wave;
  if (t >= T_TOK) return;
  const float* xr = x + (size_t)t * DMODEL;

  float acc[NEXP];
  #pragma unroll
  for (int e = 0; e < NEXP; ++e) acc[e] = 0.f;
  for (int d = lane; d < DMODEL; d += 32) {
    float xv = xr[d];
    #pragma unroll
    for (int e = 0; e < NEXP; ++e) acc[e] += xv * wr[e * DMODEL + d];
  }
  #pragma unroll
  for (int e = 0; e < NEXP; ++e) {
    #pragma unroll
    for (int off = 16; off > 0; off >>= 1) acc[e] += __shfl_xor(acc[e], off, 32);
  }
  int   be = 0;
  float bv = acc[0];
  #pragma unroll
  for (int e = 1; e < NEXP; ++e) { if (acc[e] > bv) { bv = acc[e]; be = e; } }
  float w = 1.f / (1.f + __expf(-bv));   // sigmoid gate weight

  if (lane == 0) {
    int pos = atomicAdd(&counts[be], 1);
    lists[be * T_TOK + pos] = t;
  }
  uint16_t* xbp = xb + (size_t)t * DMODEL;
  uint16_t* xsp = xs + (size_t)t * DMODEL;
  for (int d = lane; d < DMODEL; d += 32) {
    float f = xr[d];
    xbp[d] = f32_bf16(f);
    xsp[d] = f32_bf16(w * f);
  }
}

// --------------------------------------------------------------------------
// Fused gate/up GEMM + SwiGLU epilogue:  H = silu(A @ Wg^T) * (A @ Wu^T)
// A: bf16 [T,D] (rows gathered via expert token list when routed).
// Wg/Wu: bf16 [I,D] row-major (per-expert slice, pre-converted).
// Double-buffered async global->LDS staging.
// --------------------------------------------------------------------------
__global__ void __launch_bounds__(256)
swiglu_gemm(const uint16_t* __restrict__ Abf, const uint16_t* __restrict__ Wg,
            const uint16_t* __restrict__ Wu, uint16_t* __restrict__ H,
            const int* __restrict__ lists, const int* __restrict__ counts,
            int routed) {
  __shared__ uint16_t AtS[2 * BM * LDST];
  __shared__ uint16_t BgS[2 * BN * LDST];
  __shared__ uint16_t BuS[2 * BN * LDST];

  const int mtiles = T_TOK / BM;    // 128
  const int ntiles = IMODEL / BN;   // 32
  int bid = blockIdx.x;
  int e = 0, cnt = T_TOK;
  const int* list = nullptr;
  if (routed) {
    int per = mtiles * ntiles;
    e = bid / per; bid -= e * per;
    cnt = counts[e];
    list = lists + (size_t)e * T_TOK;
  }
  int mt = bid / ntiles;
  int nt = bid % ntiles;
  if (mt * BM >= cnt) return;

  const uint16_t* wg = Wg + (size_t)e * IMODEL * DMODEL;
  const uint16_t* wu = Wu + (size_t)e * IMODEL * DMODEL;

  int tid  = threadIdx.x;
  int lane = tid & 31;
  int wave = tid >> 5;
  int wm = wave >> 1;   // 0..3 : wave M offset = 32*wm
  int wn = wave & 1;    // 0..1 : wave N offset = 32*wn

  // A-stage indexing: two 16B chunks per thread, B-stage: one per matrix.
  int aidx0 = tid * 8;            // u16 index within 128x32 tile, pass 0
  int aidx1 = (256 + tid) * 8;    // pass 1
  int ar0 = aidx0 >> 5, ac0 = aidx0 & 31;
  int ar1 = aidx1 >> 5, ac1 = aidx1 & 31;
  int gm0 = mt * BM + ar0;
  int gm1 = mt * BM + ar1;
  int row0 = routed ? list[min(gm0, cnt - 1)] : gm0;
  int row1 = routed ? list[min(gm1, cnt - 1)] : gm1;
  int bidx = tid * 8;             // u16 index within 64x32 tile
  int bn = bidx >> 5, bc = bidx & 31;
  const uint16_t* gsrcBase = wg + (size_t)(nt * BN + bn) * DMODEL + bc;
  const uint16_t* usrcBase = wu + (size_t)(nt * BN + bn) * DMODEL + bc;

  auto stage = [&](int kb, int buf) {
    uint16_t* At = AtS + buf * (BM * LDST);
    uint16_t* Bg = BgS + buf * (BN * LDST);
    uint16_t* Bu = BuS + buf * (BN * LDST);
    copy16_g2lds(Abf + (size_t)row0 * DMODEL + kb + ac0, &At[ar0 * LDST + ac0]);
    copy16_g2lds(Abf + (size_t)row1 * DMODEL + kb + ac1, &At[ar1 * LDST + ac1]);
    copy16_g2lds(gsrcBase + kb, &Bg[bn * LDST + bc]);
    copy16_g2lds(usrcBase + kb, &Bu[bn * LDST + bc]);
  };

  v8f accg[2][2], accu[2][2];
  #pragma unroll
  for (int i = 0; i < 2; ++i) {
    #pragma unroll
    for (int j = 0; j < 2; ++j) { accg[i][j] = zero8(); accu[i][j] = zero8(); }
  }

  const int nk = DMODEL / BK;   // 32
  stage(0, 0);
  for (int k = 0; k < nk; ++k) {
    int cur = k & 1;
    if (k + 1 < nk) { stage((k + 1) * BK, cur ^ 1); WAIT_ASYNC(4); }
    else            { WAIT_ASYNC(0); }
    __syncthreads();

    const uint16_t* At = AtS + cur * (BM * LDST);
    const uint16_t* Bg = BgS + cur * (BN * LDST);
    const uint16_t* Bu = BuS + cur * (BN * LDST);

    // fragments (ISA 16-bit A/B layout: lane&15 = row, lane>>4 = K half)
    int khalf = (lane >> 4) * 8;
    FragBF a[2], bg[2], bu[2];
    #pragma unroll
    for (int i = 0; i < 2; ++i) {
      const uint16_t* p = &At[(wm * 32 + i * 16 + (lane & 15)) * LDST + khalf];
      a[i].q[0] = *reinterpret_cast<const uint4*>(p);       // K = k0..k0+7
      a[i].q[1] = *reinterpret_cast<const uint4*>(p + 16);  // K = k0+16..k0+23
    }
    #pragma unroll
    for (int j = 0; j < 2; ++j) {
      const uint16_t* pg = &Bg[(wn * 32 + j * 16 + (lane & 15)) * LDST + khalf];
      bg[j].q[0] = *reinterpret_cast<const uint4*>(pg);
      bg[j].q[1] = *reinterpret_cast<const uint4*>(pg + 16);
      const uint16_t* pu = &Bu[(wn * 32 + j * 16 + (lane & 15)) * LDST + khalf];
      bu[j].q[0] = *reinterpret_cast<const uint4*>(pu);
      bu[j].q[1] = *reinterpret_cast<const uint4*>(pu + 16);
    }
    #pragma unroll
    for (int i = 0; i < 2; ++i) {
      #pragma unroll
      for (int j = 0; j < 2; ++j) {
        accg[i][j] = __builtin_amdgcn_wmma_f32_16x16x32_bf16(
            false, a[i].v, false, bg[j].v, (short)0, accg[i][j], false, false);
        accu[i][j] = __builtin_amdgcn_wmma_f32_16x16x32_bf16(
            false, a[i].v, false, bu[j].v, (short)0, accu[i][j], false, false);
      }
    }
    __syncthreads();
  }

  // SwiGLU epilogue, write H in bf16
  #pragma unroll
  for (int i = 0; i < 2; ++i) {
    #pragma unroll
    for (int j = 0; j < 2; ++j) {
      int coln = nt * BN + wn * 32 + j * 16 + (lane & 15);
      #pragma unroll
      for (int p = 0; p < 8; ++p) {
        int rl = wm * 32 + i * 16 + ((lane >> 4) * 8) + p;
        int gm = mt * BM + rl;
        if (gm < cnt) {
          int row = routed ? list[gm] : gm;
          float g = accg[i][j][p];
          float u = accu[i][j][p];
          float h = g * (1.f / (1.f + __expf(-g))) * u;
          H[(size_t)row * IMODEL + coln] = f32_bf16(h);
        }
      }
    }
  }
}

// --------------------------------------------------------------------------
// Down projection:  out (+)= H @ Wd^T.   H: bf16 [T,I]; Wd: bf16 [D,I].
// Shared expert writes, routed experts accumulate (disjoint token rows).
// --------------------------------------------------------------------------
__global__ void __launch_bounds__(256)
down_gemm(const uint16_t* __restrict__ Hbf, const uint16_t* __restrict__ Wd,
          float* __restrict__ out, const int* __restrict__ lists,
          const int* __restrict__ counts, int routed) {
  __shared__ uint16_t AtS[2 * BM * LDST];
  __shared__ uint16_t BtS[2 * BN * LDST];

  const int mtiles = T_TOK / BM;    // 128
  const int ntiles = DMODEL / BN;   // 16
  int bid = blockIdx.x;
  int e = 0, cnt = T_TOK;
  const int* list = nullptr;
  if (routed) {
    int per = mtiles * ntiles;
    e = bid / per; bid -= e * per;
    cnt = counts[e];
    list = lists + (size_t)e * T_TOK;
  }
  int mt = bid / ntiles;
  int nt = bid % ntiles;
  if (mt * BM >= cnt) return;

  const uint16_t* wd = Wd + (size_t)e * DMODEL * IMODEL;

  int tid  = threadIdx.x;
  int lane = tid & 31;
  int wave = tid >> 5;
  int wm = wave >> 1;
  int wn = wave & 1;

  int aidx0 = tid * 8;
  int aidx1 = (256 + tid) * 8;
  int ar0 = aidx0 >> 5, ac0 = aidx0 & 31;
  int ar1 = aidx1 >> 5, ac1 = aidx1 & 31;
  int gm0 = mt * BM + ar0;
  int gm1 = mt * BM + ar1;
  int row0 = routed ? list[min(gm0, cnt - 1)] : gm0;
  int row1 = routed ? list[min(gm1, cnt - 1)] : gm1;
  int bidx = tid * 8;
  int bn = bidx >> 5, bc = bidx & 31;
  const uint16_t* dsrcBase = wd + (size_t)(nt * BN + bn) * IMODEL + bc;

  auto stage = [&](int kb, int buf) {
    uint16_t* At = AtS + buf * (BM * LDST);
    uint16_t* Bt = BtS + buf * (BN * LDST);
    copy16_g2lds(Hbf + (size_t)row0 * IMODEL + kb + ac0, &At[ar0 * LDST + ac0]);
    copy16_g2lds(Hbf + (size_t)row1 * IMODEL + kb + ac1, &At[ar1 * LDST + ac1]);
    copy16_g2lds(dsrcBase + kb, &Bt[bn * LDST + bc]);
  };

  v8f acc[2][2];
  #pragma unroll
  for (int i = 0; i < 2; ++i) {
    #pragma unroll
    for (int j = 0; j < 2; ++j) acc[i][j] = zero8();
  }

  const int nk = IMODEL / BK;   // 64
  stage(0, 0);
  for (int k = 0; k < nk; ++k) {
    int cur = k & 1;
    if (k + 1 < nk) { stage((k + 1) * BK, cur ^ 1); WAIT_ASYNC(3); }
    else            { WAIT_ASYNC(0); }
    __syncthreads();

    const uint16_t* At = AtS + cur * (BM * LDST);
    const uint16_t* Bt = BtS + cur * (BN * LDST);

    int khalf = (lane >> 4) * 8;
    FragBF a[2], b[2];
    #pragma unroll
    for (int i = 0; i < 2; ++i) {
      const uint16_t* p = &At[(wm * 32 + i * 16 + (lane & 15)) * LDST + khalf];
      a[i].q[0] = *reinterpret_cast<const uint4*>(p);
      a[i].q[1] = *reinterpret_cast<const uint4*>(p + 16);
    }
    #pragma unroll
    for (int j = 0; j < 2; ++j) {
      const uint16_t* p = &Bt[(wn * 32 + j * 16 + (lane & 15)) * LDST + khalf];
      b[j].q[0] = *reinterpret_cast<const uint4*>(p);
      b[j].q[1] = *reinterpret_cast<const uint4*>(p + 16);
    }
    #pragma unroll
    for (int i = 0; i < 2; ++i) {
      #pragma unroll
      for (int j = 0; j < 2; ++j) {
        acc[i][j] = __builtin_amdgcn_wmma_f32_16x16x32_bf16(
            false, a[i].v, false, b[j].v, (short)0, acc[i][j], false, false);
      }
    }
    __syncthreads();
  }

  #pragma unroll
  for (int i = 0; i < 2; ++i) {
    #pragma unroll
    for (int j = 0; j < 2; ++j) {
      int coln = nt * BN + wn * 32 + j * 16 + (lane & 15);
      #pragma unroll
      for (int p = 0; p < 8; ++p) {
        int rl = wm * 32 + i * 16 + ((lane >> 4) * 8) + p;
        int gm = mt * BM + rl;
        if (gm < cnt) {
          int row = routed ? list[gm] : gm;
          size_t idx = (size_t)row * DMODEL + coln;
          float v = acc[i][j][p];
          if (routed) out[idx] = out[idx] + v;   // disjoint rows across experts
          else        out[idx] = v;              // shared expert initializes
        }
      }
    }
  }
}

// --------------------------------------------------------------------------
extern "C" void kernel_launch(void* const* d_in, const int* in_sizes, int n_in,
                              void* d_out, int out_size, void* d_ws, size_t ws_size,
                              hipStream_t stream) {
  const float* x   = (const float*)d_in[0];   // [T, D]
  const float* wr  = (const float*)d_in[1];   // [E, D]
  const float* w1  = (const float*)d_in[2];   // [E, I, D]
  const float* w3  = (const float*)d_in[3];   // [E, I, D]
  const float* w2  = (const float*)d_in[4];   // [E, D, I]
  const float* wsg = (const float*)d_in[5];   // [I, D]
  const float* wsu = (const float*)d_in[6];   // [I, D]
  const float* wsd = (const float*)d_in[7];   // [D, I]
  float* out = (float*)d_out;                 // [T, D] f32

  uint8_t* ws = (uint8_t*)d_ws;
  size_t off = 0;
  auto wsalloc = [&](size_t bytes) -> void* {
    void* p = ws + off;
    off += (bytes + 255) & ~(size_t)255;
    return p;
  };
  const size_t nW  = (size_t)NEXP * IMODEL * DMODEL;  // per routed tensor
  const size_t nWs = (size_t)IMODEL * DMODEL;         // per shared tensor
  uint16_t* xb     = (uint16_t*)wsalloc((size_t)T_TOK * DMODEL * 2);
  uint16_t* xs     = (uint16_t*)wsalloc((size_t)T_TOK * DMODEL * 2);
  uint16_t* Hbuf   = (uint16_t*)wsalloc((size_t)T_TOK * IMODEL * 2);
  int*      counts = (int*)wsalloc(NEXP * sizeof(int));
  int*      lists  = (int*)wsalloc((size_t)NEXP * T_TOK * sizeof(int));
  uint16_t* w1b    = (uint16_t*)wsalloc(nW * 2);
  uint16_t* w3b    = (uint16_t*)wsalloc(nW * 2);
  uint16_t* w2b    = (uint16_t*)wsalloc(nW * 2);
  uint16_t* wsgb   = (uint16_t*)wsalloc(nWs * 2);
  uint16_t* wsub   = (uint16_t*)wsalloc(nWs * 2);
  uint16_t* wsdb   = (uint16_t*)wsalloc(nWs * 2);

  (void)hipMemsetAsync(counts, 0, NEXP * sizeof(int), stream);

  // weight pre-conversion to bf16 (enables pure-copy async LDS staging)
  const int CB = 256 * 8;
  cvt_bf16_kernel<<<(int)(nW  / CB), 256, 0, stream>>>(w1,  w1b,  (int)nW);
  cvt_bf16_kernel<<<(int)(nW  / CB), 256, 0, stream>>>(w3,  w3b,  (int)nW);
  cvt_bf16_kernel<<<(int)(nW  / CB), 256, 0, stream>>>(w2,  w2b,  (int)nW);
  cvt_bf16_kernel<<<(int)(nWs / CB), 256, 0, stream>>>(wsg, wsgb, (int)nWs);
  cvt_bf16_kernel<<<(int)(nWs / CB), 256, 0, stream>>>(wsu, wsub, (int)nWs);
  cvt_bf16_kernel<<<(int)(nWs / CB), 256, 0, stream>>>(wsd, wsdb, (int)nWs);

  router_kernel<<<T_TOK / 8, 256, 0, stream>>>(x, wr, xb, xs, counts, lists);

  const int mt  = T_TOK / BM;     // 128
  const int nt1 = IMODEL / BN;    // 32
  const int nt2 = DMODEL / BN;    // 16

  // shared expert (writes out)
  swiglu_gemm<<<mt * nt1, 256, 0, stream>>>(xb, wsgb, wsub, Hbuf, nullptr, nullptr, 0);
  down_gemm  <<<mt * nt2, 256, 0, stream>>>(Hbuf, wsdb, out, nullptr, nullptr, 0);

  // routed experts (accumulate into out); worst-case grid, early-exit on count
  swiglu_gemm<<<NEXP * mt * nt1, 256, 0, stream>>>(xs, w1b, w3b, Hbuf, lists, counts, 1);
  down_gemm  <<<NEXP * mt * nt2, 256, 0, stream>>>(Hbuf, w2b, out, lists, counts, 1);
}
